// SimGCL_56676388438555
// MI455X (gfx1250) — matrix operand order repaired and checked by reading, not compiled, not run on hardware
//
#include <hip/hip_runtime.h>
#include <stdint.h>

#define N_USERS   100000
#define N_ITEMS   200000
#define N_NODES   (N_USERS + N_ITEMS)
#define D         64
#define N_LAYERS  3
#define EPS_F     0.1f
#define TOTAL_ELEMS ((size_t)N_NODES * D)       /* 19,200,000 */
#define HALF_CNT  9600000u                      /* TOTAL_ELEMS / 2 */
#define TROWS     64                            /* rows per update block */

typedef __attribute__((ext_vector_type(16))) _Float16 v16h;
typedef __attribute__((ext_vector_type(8)))  float    v8f;

/* ------------------------------------------------------------------ */
/* Threefry-2x32 (exactly matching jax._src.prng.threefry2x32)         */
/* ------------------------------------------------------------------ */
__host__ __device__ __forceinline__ uint32_t rotl32(uint32_t x, uint32_t r) {
    return (x << r) | (x >> (32u - r));
}

__host__ __device__ __forceinline__ void threefry2x32(uint32_t k0, uint32_t k1,
                                                      uint32_t x0, uint32_t x1,
                                                      uint32_t& y0, uint32_t& y1) {
    uint32_t ks2 = k0 ^ k1 ^ 0x1BD11BDAu;
    x0 += k0; x1 += k1;
    /* group 1: 13,15,26,6 */
    x0 += x1; x1 = rotl32(x1, 13); x1 ^= x0;
    x0 += x1; x1 = rotl32(x1, 15); x1 ^= x0;
    x0 += x1; x1 = rotl32(x1, 26); x1 ^= x0;
    x0 += x1; x1 = rotl32(x1,  6); x1 ^= x0;
    x0 += k1; x1 += ks2 + 1u;
    /* group 2: 17,29,16,24 */
    x0 += x1; x1 = rotl32(x1, 17); x1 ^= x0;
    x0 += x1; x1 = rotl32(x1, 29); x1 ^= x0;
    x0 += x1; x1 = rotl32(x1, 16); x1 ^= x0;
    x0 += x1; x1 = rotl32(x1, 24); x1 ^= x0;
    x0 += ks2; x1 += k0 + 2u;
    /* group 3: 13,15,26,6 */
    x0 += x1; x1 = rotl32(x1, 13); x1 ^= x0;
    x0 += x1; x1 = rotl32(x1, 15); x1 ^= x0;
    x0 += x1; x1 = rotl32(x1, 26); x1 ^= x0;
    x0 += x1; x1 = rotl32(x1,  6); x1 ^= x0;
    x0 += k0; x1 += k1 + 3u;
    /* group 4: 17,29,16,24 */
    x0 += x1; x1 = rotl32(x1, 17); x1 ^= x0;
    x0 += x1; x1 = rotl32(x1, 29); x1 ^= x0;
    x0 += x1; x1 = rotl32(x1, 16); x1 ^= x0;
    x0 += x1; x1 = rotl32(x1, 24); x1 ^= x0;
    x0 += k1; x1 += ks2 + 4u;
    /* group 5: 13,15,26,6 */
    x0 += x1; x1 = rotl32(x1, 13); x1 ^= x0;
    x0 += x1; x1 = rotl32(x1, 15); x1 ^= x0;
    x0 += x1; x1 = rotl32(x1, 26); x1 ^= x0;
    x0 += x1; x1 = rotl32(x1,  6); x1 ^= x0;
    x0 += ks2; x1 += k0 + 5u;
    y0 = x0; y1 = x1;
}

/* uniform [0,1) for flat element i of a (N_NODES*D,) jax.random.uniform draw */
__device__ __forceinline__ float tf_uniform(uint32_t k0, uint32_t k1, uint32_t i) {
    bool lo = i < HALF_CNT;
    uint32_t j = lo ? i : (i - HALF_CNT);
    uint32_t y0, y1;
    threefry2x32(k0, k1, j, j + HALF_CNT, y0, y1);
    uint32_t bits = lo ? y0 : y1;
    return __uint_as_float((bits >> 9) | 0x3f800000u) - 1.0f;
}

/* ------------------------------------------------------------------ */
/* Kernels                                                             */
/* ------------------------------------------------------------------ */

/* x = acc = concat(user_emb, item_emb) */
__global__ void simgcl_init(const float* __restrict__ ue, const float* __restrict__ ie,
                            float* __restrict__ x, float* __restrict__ acc) {
    size_t i = (size_t)blockIdx.x * blockDim.x + threadIdx.x;
    size_t stride = (size_t)gridDim.x * blockDim.x;
    const size_t ucut = (size_t)N_USERS * D;
    for (; i < TOTAL_ELEMS; i += stride) {
        float v = (i < ucut) ? ue[i] : ie[i - ucut];
        x[i] = v;
        acc[i] = v;
    }
}

/* y += A x : COO scatter; 16 lanes per nnz, float4 per lane */
__global__ void simgcl_spmm(const int* __restrict__ rows, const int* __restrict__ cols,
                            const float* __restrict__ vals, const float* __restrict__ x,
                            float* __restrict__ y, int nnz) {
    const int sub = threadIdx.x & 15;
    long long g = (long long)blockIdx.x * blockDim.x + threadIdx.x;
    long long e = g >> 4;
    const long long stride = ((long long)gridDim.x * blockDim.x) >> 4;
    for (; e < nnz; e += stride) {
        if (e + stride < nnz) {                 /* stream-ahead prefetch of index/value arrays */
            __builtin_prefetch(cols + e + stride, 0, 0);
            __builtin_prefetch(vals + e + stride, 0, 0);
        }
        const int r = rows[e];
        const int c = cols[e];
        const float v = vals[e];
        const float4 xv = *(const float4*)(x + (size_t)c * D + sub * 4);
        float* yp = y + (size_t)r * D + sub * 4;
        unsafeAtomicAdd(yp + 0, v * xv.x);
        unsafeAtomicAdd(yp + 1, v * xv.y);
        unsafeAtomicAdd(yp + 2, v * xv.z);
        unsafeAtomicAdd(yp + 3, v * xv.w);
    }
}

/* noise injection: ego = y + sign(y)*normalize(noise)*EPS ; x=ego ; acc+=ego
   - y tile staged into LDS with async global->LDS loads (ASYNCcnt),
     overlapped with the threefry PRNG phase
   - row L2-norms via WMMA Gram-matrix diagonal (16-row tiles per wave) */
__global__ __launch_bounds__(128) void simgcl_update(const float* __restrict__ y,
                                                     float* __restrict__ x,
                                                     float* __restrict__ acc,
                                                     uint32_t k0, uint32_t k1) {
    __shared__ float s_noise[TROWS * 64];
    __shared__ float s_y[TROWS * 64];
    __shared__ float s_norm2[TROWS];

    const int t = threadIdx.x;                  /* 0..127, 4 waves */
    const int blockRow = blockIdx.x * TROWS;

    /* phase 0: kick off async global->LDS copy of the y tile (B128 per lane).
       TROWS*64 floats = TROWS*16 float4s; 128 threads -> 8 issues/thread. */
    const uint32_t s_y_base = (uint32_t)(uintptr_t)s_y;
    for (int q = t; q < TROWS * 16; q += 128) {
        const int row = q >> 4;                 /* 16 float4 per row */
        const int grow = blockRow + row;
        if (grow < N_NODES) {                   /* per-lane EXEC masking of the async op */
            const uint64_t ga = (uint64_t)(uintptr_t)(y + (size_t)grow * D + (size_t)(q & 15) * 4);
            const uint32_t la = s_y_base + (uint32_t)q * 16u;
            asm volatile("global_load_async_to_lds_b128 %0, %1, off"
                         :: "v"(la), "v"(ga) : "memory");
        }
    }

    /* phase 1: generate noise tile (threefry) into LDS -- hides the async loads */
    for (int idx = t; idx < TROWS * 64; idx += 128) {
        const int row = idx >> 6, col = idx & 63;
        const int grow = blockRow + row;
        float u = 0.0f;
        if (grow < N_NODES) u = tf_uniform(k0, k1, (uint32_t)grow * 64u + (uint32_t)col);
        s_noise[idx] = u;
    }

    /* drain async copies, then rendezvous */
#if __has_builtin(__builtin_amdgcn_s_wait_asynccnt)
    __builtin_amdgcn_s_wait_asynccnt(0);
#else
    asm volatile("s_wait_asynccnt 0" ::: "memory");
#endif
    __syncthreads();

    /* phase 2: per-wave 16-row Gram diagonal via v_wmma_f32_16x16x32_f16.
       Uniform control flow here: EXEC all-1s for the WMMA. */
    const int lane = t & 31;
    const int wv   = t >> 5;                    /* 0..3 */
    const int R    = wv * 16;
    const int mA   = lane & 15;                 /* A row == B column for this lane */
    const int hi   = lane >> 4;

    v8f c = {};
#pragma unroll
    for (int kb = 0; kb < 64; kb += 32) {
        v16h a, b;
#pragma unroll
        for (int h = 0; h < 16; ++h) {
            const int v   = h >> 1;
            const int off = ((v < 4) ? (2 * v) : (16 + 2 * (v - 4))) + (h & 1);
            const int kA  = kb + off + (hi ? 8 : 0);
            const int kB  = kb + off + (hi ? 8 : 0);
            a[h] = (_Float16)s_noise[(R + mA) * 64 + kA];
            b[h] = (_Float16)s_noise[(R + mA) * 64 + kB];
        }
        c = __builtin_amdgcn_wmma_f32_16x16x32_f16(false, a, false, b,
                                                   (short)0, c, false, false);
    }
    /* diagonal: M=N=m. m<8 -> c[m]@lane m ; m>=8 -> c[m-8]@lane m+16 */
    float dv;
    switch (lane & 7) {
        case 0: dv = c[0]; break;  case 1: dv = c[1]; break;
        case 2: dv = c[2]; break;  case 3: dv = c[3]; break;
        case 4: dv = c[4]; break;  case 5: dv = c[5]; break;
        case 6: dv = c[6]; break;  default: dv = c[7]; break;
    }
    if (lane < 8)        s_norm2[R + lane] = dv;
    else if (lane >= 24) s_norm2[R + lane - 16] = dv;
    __syncthreads();

    /* phase 3: sign-noise injection + accumulate (y read from LDS stage) */
    for (int idx = t; idx < TROWS * 64; idx += 128) {
        const int row = idx >> 6, col = idx & 63;
        const int grow = blockRow + row;
        if (grow < N_NODES) {
            const size_t gi = (size_t)grow * D + col;
            const float yv  = s_y[idx];
            const float inv = rsqrtf(s_norm2[row]);
            const float nz  = s_noise[idx] * inv;
            const float sg  = (yv > 0.0f) ? 1.0f : ((yv < 0.0f) ? -1.0f : 0.0f);
            const float ego = yv + sg * nz * EPS_F;
            x[gi] = ego;
            acc[gi] += ego;
        }
    }
}

/* mean over (n_layers+1) embeddings */
__global__ void simgcl_scale(float* __restrict__ acc) {
    size_t i = (size_t)blockIdx.x * blockDim.x + threadIdx.x;
    size_t stride = (size_t)gridDim.x * blockDim.x;
    for (; i < TOTAL_ELEMS; i += stride)
        acc[i] *= 0.25f;
}

/* ------------------------------------------------------------------ */
extern "C" void kernel_launch(void* const* d_in, const int* in_sizes, int n_in,
                              void* d_out, int out_size, void* d_ws, size_t ws_size,
                              hipStream_t stream) {
    (void)n_in; (void)out_size; (void)ws_size;
    const int*   rows = (const int*)d_in[0];
    const int*   cols = (const int*)d_in[1];
    const float* vals = (const float*)d_in[2];
    const float* uemb = (const float*)d_in[3];
    const float* iemb = (const float*)d_in[4];
    const int    nnz  = in_sizes[0];

    float* acc = (float*)d_out;                 /* running sum, scaled at the end */
    float* x   = (float*)d_ws;                  /* current ego      [N, D] */
    float* y   = x + TOTAL_ELEMS;               /* spmm destination [N, D] */

    /* layer keys: fold_in(key(42), k) == threefry2x32(key=(0,42), count=(0,k)) */
    uint32_t lk0[N_LAYERS], lk1[N_LAYERS];
    for (int k = 0; k < N_LAYERS; ++k)
        threefry2x32(0u, 42u, 0u, (uint32_t)k, lk0[k], lk1[k]);

    const int ew_blocks   = (int)((TOTAL_ELEMS + 255) / 256);   /* 75000 */
    const int upd_blocks  = (N_NODES + TROWS - 1) / TROWS;      /* 4688 */
    const int spmm_blocks = 8192;

    simgcl_init<<<ew_blocks, 256, 0, stream>>>(uemb, iemb, x, acc);

    for (int k = 0; k < N_LAYERS; ++k) {
        hipMemsetAsync(y, 0, TOTAL_ELEMS * sizeof(float), stream);
        simgcl_spmm<<<spmm_blocks, 256, 0, stream>>>(rows, cols, vals, x, y, nnz);
        simgcl_update<<<upd_blocks, 128, 0, stream>>>(y, x, acc, lk0[k], lk1[k]);
    }

    simgcl_scale<<<ew_blocks, 256, 0, stream>>>(acc);
}
